// LofteyModel_24300924961530
// MI455X (gfx1250) — matrix operand order repaired and checked by reading, not compiled, not run on hardware
//
#include <hip/hip_runtime.h>
#include <hip/hip_bf16.h>

// ---------------- model dims ----------------
constexpr int cV  = 32000, cH = 1024, cI = 2048, cL = 4;
constexpr int cNH = 16, cNKV = 8, cHD = 64, cE = 4;
constexpr int cB  = 2, cS = 1024, cT = cB * cS;
constexpr int cQD = cNH * cHD;    // 1024
constexpr int cKD = cNKV * cHD;   // 512

typedef unsigned short u16;
typedef __attribute__((ext_vector_type(16))) __bf16 v16bf;
typedef __attribute__((ext_vector_type(8)))  float  v8f;

// ---------------- bf16 helpers ----------------
__device__ __forceinline__ u16 f2bfu(float f) {
  unsigned u = __builtin_bit_cast(unsigned, f);
  unsigned r = u + 0x7FFFu + ((u >> 16) & 1u);   // RNE
  return (u16)(r >> 16);
}
__device__ __forceinline__ float bfu2f(u16 s) {
  return __builtin_bit_cast(float, (unsigned)s << 16);
}
struct U32x8 { uint4 a, b; };
__device__ __forceinline__ v16bf ld_frag(const u16* p0, const u16* p1) {
  U32x8 t;
  t.a = *(const uint4*)p0;
  t.b = *(const uint4*)p1;
  return __builtin_bit_cast(v16bf, t);
}
__device__ __forceinline__ v8f wmma_bf16(v16bf a, v16bf b, v8f c) {
  return __builtin_amdgcn_wmma_f32_16x16x32_bf16(false, a, false, b, (short)0, c,
                                                 false, false);
}

// ---------------- async global->LDS staging (CDNA5), guarded ----------------
#if __has_builtin(__builtin_amdgcn_s_wait_asynccnt)
#define WAIT_ASYNC0() __builtin_amdgcn_s_wait_asynccnt(0)
#define WAIT_ASYNC3() __builtin_amdgcn_s_wait_asynccnt(3)
#else
#define WAIT_ASYNC0() asm volatile("s_wait_asynccnt 0x0" ::: "memory")
#define WAIT_ASYNC3() asm volatile("s_wait_asynccnt 0x3" ::: "memory")
#endif

typedef int i4v __attribute__((vector_size(16)));
typedef __attribute__((address_space(1))) i4v* gp_i4;   // global b128 ptr
typedef __attribute__((address_space(3))) i4v* lp_i4;   // LDS b128 ptr

#if __has_builtin(__builtin_amdgcn_global_load_async_to_lds_b128)
#define ASYNC_CP16(gsrc, ldst)                                                  \
  __builtin_amdgcn_global_load_async_to_lds_b128((gp_i4)(gsrc), (lp_i4)(ldst),  \
                                                 0, 0)
#define HAVE_ASYNC 1
#else
#define ASYNC_CP16(gsrc, ldst)                                                  \
  do { *(uint4*)(ldst) = *(const uint4*)(gsrc); } while (0)
#define HAVE_ASYNC 0
#endif

// ---------------- fp32 -> bf16, K-packed tiling: dst[k/32][n][k%32] ----------------
__global__ void __launch_bounds__(256)
cvt_pack_bf16(const float* __restrict__ src, u16* __restrict__ dst, int Kdim, int N) {
  const int gid = blockIdx.x * 256 + threadIdx.x;
  const int n4  = N >> 2;
  if (gid >= Kdim * n4) return;
  const int k    = gid / n4;
  const int col  = (gid - k * n4) * 4;
  const float4 f = *(const float4*)(src + (size_t)k * N + col);
  u16* d = dst + ((size_t)(k >> 5) * N) * 32 + (k & 31);
  d[(size_t)(col + 0) * 32] = f2bfu(f.x);
  d[(size_t)(col + 1) * 32] = f2bfu(f.y);
  d[(size_t)(col + 2) * 32] = f2bfu(f.z);
  d[(size_t)(col + 3) * 32] = f2bfu(f.w);
}

// ---------------- embedding gather (fp32 h) ----------------
__global__ void __launch_bounds__(256)
embed_kernel(const int* __restrict__ ids, const float* __restrict__ E,
             float* __restrict__ H) {
  const int t  = blockIdx.x;
  const int id = ids[t];
  const float* src = E + (size_t)id * cH;
  float*       dst = H + (size_t)t  * cH;
  for (int j = threadIdx.x; j < cH; j += 256) dst[j] = src[j];
}

// ---------------- RMSNorm: fp32 in -> bf16 out ----------------
__global__ void __launch_bounds__(256)
rmsnorm_kernel(const float* __restrict__ X, const float* __restrict__ W,
               u16* __restrict__ Y) {
  __shared__ float red[8];
  const int t = blockIdx.x;
  const float* x = X + (size_t)t * cH;
  float ss = 0.f;
  for (int j = threadIdx.x; j < cH; j += 256) { float v = x[j]; ss += v * v; }
#pragma unroll
  for (int m = 1; m < 32; m <<= 1) ss += __shfl_xor(ss, m);
  if ((threadIdx.x & 31) == 0) red[threadIdx.x >> 5] = ss;
  __syncthreads();
  float tot = 0.f;
#pragma unroll
  for (int i = 0; i < 8; i++) tot += red[i];
  const float rs = rsqrtf(tot / (float)cH + 1e-6f);
  u16* y = Y + (size_t)t * cH;
  for (int j = threadIdx.x; j < cH; j += 256) y[j] = f2bfu(W[j] * x[j] * rs);
}

// ---------------- MoE gating from bf16 x ----------------
__global__ void __launch_bounds__(128)
gating_kernel(const u16* __restrict__ X, const float* __restrict__ G,
              float* __restrict__ comb) {
  const int t  = blockIdx.x * 4 + (threadIdx.x >> 5);
  const int ln = threadIdx.x & 31;
  const u16* x = X + (size_t)t * cH;
  float z[cE] = {0.f, 0.f, 0.f, 0.f};
  for (int i = ln; i < cH; i += 32) {
    const float xv = bfu2f(x[i]);
#pragma unroll
    for (int e = 0; e < cE; e++) z[e] += xv * G[(size_t)i * cE + e];
  }
#pragma unroll
  for (int m = 1; m < 32; m <<= 1) {
#pragma unroll
    for (int e = 0; e < cE; e++) z[e] += __shfl_xor(z[e], m);
  }
  float mx = fmaxf(fmaxf(z[0], z[1]), fmaxf(z[2], z[3]));
  float p[cE], sum = 0.f;
#pragma unroll
  for (int e = 0; e < cE; e++) { p[e] = __expf(z[e] - mx); sum += p[e]; }
#pragma unroll
  for (int e = 0; e < cE; e++) p[e] /= sum;
  int i1 = 0;
#pragma unroll
  for (int e = 1; e < cE; e++) if (p[e] > p[i1]) i1 = e;
  int i2 = -1;
#pragma unroll
  for (int e = 0; e < cE; e++)
    if (e != i1 && (i2 < 0 || p[e] > p[i2])) i2 = e;
  if (ln == 0) {
#pragma unroll
    for (int e = 0; e < cE; e++)
      comb[(size_t)t * cE + e] = (e == i1 || e == i2) ? p[e] : 0.f;
  }
}

// ---------------- double-buffered LDS-staged bf16 WMMA GEMM ----------------
// A row-major bf16 [M,K]; B K-packed bf16 Bt[k/32][N][k%32]; tiles 64x128x32.
// 8 waves; each wave: 16 rows x 64 cols = one A-frag reused across 4 WMMAs.
// mode: 0 Cf store, 1 Cf +=, 2 Cb = bf16(silu(aux[m,n]) * acc),
//       3 Cf += scale[m*sstride]*acc, 4 Cb store, 5 Cb transposed store (ld=tld)
__global__ void __launch_bounds__(256)
gemm_bf16_lds(const u16* __restrict__ A, int lda,
              const u16* __restrict__ B, int ldb,
              float* __restrict__ Cf, u16* __restrict__ Cb, int ldc,
              int Kdim, int mode,
              const float* __restrict__ aux, int aux_ld,
              const float* __restrict__ scale, int sstride, int tld) {
  __shared__ u16 As[2][64 * 32];    // 2 x 4 KB, row-major [m][k]
  __shared__ u16 Bs[2][128 * 32];   // 2 x 8 KB, packed  [n][k]
  const int tid  = threadIdx.x;
  const int lane = tid & 31;
  const int wave = tid >> 5;
  const int hl   = lane >> 4;
  const int lr   = lane & 15;
  const int wr   = wave >> 1;   // 0..3 row tile (16 rows)
  const int wc   = wave & 1;    // 0..1 col half (64 cols)
  const int m0 = blockIdx.y * 64;
  const int n0 = blockIdx.x * 128;

  // staging coordinates: 3 b128 async ops per thread per k-step
  const int ar  = tid >> 2, seg = (tid & 3) * 8;   // A: 64 rows x 32
  const int br0 = tid >> 2, br1 = br0 + 64;        // B: 128 n-rows x 32

  auto stage = [&](int bi, int k0) {
    ASYNC_CP16(A + (size_t)(m0 + ar) * lda + k0 + seg, &As[bi][ar * 32 + seg]);
    const u16* bb = B + ((size_t)(k0 >> 5) * ldb + n0) * 32 + seg;
    ASYNC_CP16(bb + (size_t)br0 * 32, &Bs[bi][br0 * 32 + seg]);
    ASYNC_CP16(bb + (size_t)br1 * 32, &Bs[bi][br1 * 32 + seg]);
  };

  v8f acc[4] = {};
  stage(0, 0);
  int buf = 0;
  for (int k0 = 0; k0 < Kdim; k0 += 32) {
    const bool hasNext = (k0 + 32 < Kdim);
    if (hasNext) {
      stage(buf ^ 1, k0 + 32);   // prefetch next tile, overlapped with wait+compute
      WAIT_ASYNC3();             // only the current buffer's 3 ops must retire
    } else {
      WAIT_ASYNC0();
    }
    __syncthreads();
    const u16* Ab = &As[buf][(wr * 16 + lr) * 32 + hl * 8];
    const v16bf a = ld_frag(Ab, Ab + 16);
#pragma unroll
    for (int g = 0; g < 4; g++) {
      const u16* Bb = &Bs[buf][(wc * 64 + g * 16 + lr) * 32 + hl * 8];
      const v16bf b = ld_frag(Bb, Bb + 16);
      acc[g] = wmma_bf16(a, b, acc[g]);
    }
    __syncthreads();
    buf ^= 1;
  }
#pragma unroll
  for (int g = 0; g < 4; g++) {
#pragma unroll
    for (int r = 0; r < 8; r++) {
      const int m = m0 + wr * 16 + hl * 8 + r;
      const int n = n0 + wc * 64 + g * 16 + lr;
      const size_t idx = (size_t)m * ldc + n;
      const float v = acc[g][r];
      if (mode == 0)      Cf[idx] = v;
      else if (mode == 1) Cf[idx] += v;
      else if (mode == 2) {
        const float gg = aux[(size_t)m * aux_ld + n];
        Cb[idx] = f2bfu((gg / (1.f + __expf(-gg))) * v);   // silu(g)*u -> bf16
      } else if (mode == 3) Cf[idx] += scale[(size_t)m * sstride] * v;
      else if (mode == 4) Cb[idx] = f2bfu(v);
      else                Cb[(size_t)n * tld + m] = f2bfu(v);   // transposed
    }
  }
}

// ---------------- fused causal attention (flash-style WMMA, bf16 IO) ----------------
// Q:[T,1024]; K:[T,512]; Vt:[512,T] transposed; O:[T,1024]. GQA rep=2.
__global__ void __launch_bounds__(128)
attn_fused(const u16* __restrict__ Q, const u16* __restrict__ Kb,
           const u16* __restrict__ Vt, u16* __restrict__ O) {
  __shared__ u16 pstage[4][16 * 32];
  const int lane = threadIdx.x & 31;
  const int wave = threadIdx.x >> 5;
  const int hl   = lane >> 4;
  const int lr   = lane & 15;
  const int bh   = blockIdx.x >> 4;
  const int qblk = blockIdx.x & 15;
  const int b    = bh >> 4;              // NH == 16
  const int hh   = bh & 15;
  const int kvh  = hh >> 1;              // rep = 2
  const int qbase = qblk * 64 + wave * 16;
  const int trow  = b * cS + qbase;
  const float sc = 0.125f;               // 1/sqrt(64)

  v16bf qa[2];
  {
    const u16* qp = Q + (size_t)(trow + lr) * cQD + hh * cHD;
#pragma unroll
    for (int c = 0; c < 2; c++) {
      const u16* p = qp + c * 32 + hl * 8;
      qa[c] = ld_frag(p, p + 16);
    }
  }

  float mrow[8], lrow[8];
  v8f oacc[4] = {};
#pragma unroll
  for (int r = 0; r < 8; r++) { mrow[r] = -1e30f; lrow[r] = 0.f; }

  const int qend = qbase + 15;
  for (int jb = 0; jb < cS && jb <= qend; jb += 32) {
    v8f sg[2];
#pragma unroll
    for (int g = 0; g < 2; g++) {
      const u16* kp = Kb + (size_t)(b * cS + jb + g * 16 + lr) * cKD + kvh * cHD + hl * 8;
      const v16bf bk0 = ld_frag(kp, kp + 16);
      const v16bf bk1 = ld_frag(kp + 32, kp + 48);
      v8f s = {};
      s = wmma_bf16(qa[0], bk0, s);
      s = wmma_bf16(qa[1], bk1, s);
      sg[g] = s;
    }
#pragma unroll
    for (int r = 0; r < 8; r++) {
      const int qi  = qbase + hl * 8 + r;
      float v0 = ((jb + lr)      > qi) ? -3.0e38f : sg[0][r] * sc;
      float v1 = ((jb + 16 + lr) > qi) ? -3.0e38f : sg[1][r] * sc;
      float rmax = fmaxf(v0, v1);
#pragma unroll
      for (int m = 1; m < 16; m <<= 1) rmax = fmaxf(rmax, __shfl_xor(rmax, m));
      const float mnew  = fmaxf(mrow[r], rmax);
      const float alpha = __expf(mrow[r] - mnew);
      const float e0 = __expf(v0 - mnew);
      const float e1 = __expf(v1 - mnew);
      float rsum = e0 + e1;
#pragma unroll
      for (int m = 1; m < 16; m <<= 1) rsum += __shfl_xor(rsum, m);
      lrow[r] = lrow[r] * alpha + rsum;
      mrow[r] = mnew;
#pragma unroll
      for (int c = 0; c < 4; c++) oacc[c][r] *= alpha;
      const int row = hl * 8 + r;
      pstage[wave][row * 32 + lr]      = f2bfu(e0);
      pstage[wave][row * 32 + 16 + lr] = f2bfu(e1);
    }
    __builtin_amdgcn_wave_barrier();
    const u16* ps = &pstage[wave][lr * 32 + hl * 8];
    const v16bf pa = ld_frag(ps, ps + 16);
    __builtin_amdgcn_wave_barrier();
#pragma unroll
    for (int c = 0; c < 4; c++) {
      const u16* vp = Vt + (size_t)(kvh * cHD + c * 16 + lr) * cT + b * cS + jb + hl * 8;
      const v16bf bv = ld_frag(vp, vp + 16);
      oacc[c] = wmma_bf16(pa, bv, oacc[c]);
    }
  }
#pragma unroll
  for (int c = 0; c < 4; c++) {
#pragma unroll
    for (int r = 0; r < 8; r++) {
      const int t = trow + hl * 8 + r;
      O[(size_t)t * cQD + hh * cHD + c * 16 + lr] = f2bfu(oacc[c][r] / lrow[r]);
    }
  }
}

// ---------------- orchestration ----------------
extern "C" void kernel_launch(void* const* d_in, const int* in_sizes, int n_in,
                              void* d_out, int out_size, void* d_ws, size_t ws_size,
                              hipStream_t stream) {
  (void)in_sizes; (void)n_in; (void)out_size; (void)ws_size;
  const int*   ids    = (const int*)  d_in[0];
  const float* embed  = (const float*)d_in[1];
  const float* Wq     = (const float*)d_in[2];
  const float* Wk     = (const float*)d_in[3];
  const float* Wv     = (const float*)d_in[4];
  const float* Wo     = (const float*)d_in[5];
  const float* ln1    = (const float*)d_in[6];
  const float* ln2    = (const float*)d_in[7];
  const float* gate   = (const float*)d_in[8];
  const float* Wg     = (const float*)d_in[9];
  const float* Wu     = (const float*)d_in[10];
  const float* Wd     = (const float*)d_in[11];
  const float* norm_w = (const float*)d_in[12];
  const float* lmh    = (const float*)d_in[13];
  float* out = (float*)d_out;

  // ---- workspace carve-up ----
  char* ws = (char*)d_ws;
  size_t off = 0;
  auto carve = [&](size_t bytes) { void* p = ws + off; off += (bytes + 255) & ~(size_t)255; return p; };
  float* h    = (float*)carve((size_t)cT * cH * 4);
  u16*   xb   = (u16*)  carve((size_t)cT * cH * 2);
  u16*   qb   = (u16*)  carve((size_t)cT * cQD * 2);
  u16*   kb   = (u16*)  carve((size_t)cT * cKD * 2);
  u16*   vtb  = (u16*)  carve((size_t)cKD * cT * 2);   // transposed V
  u16*   ob   = (u16*)  carve((size_t)cT * cQD * 2);
  float* gbuf = (float*)carve((size_t)cT * cI * 4);
  u16*   gub  = (u16*)  carve((size_t)cT * cI * 2);
  float* comb = (float*)carve((size_t)cT * cE * 4);
  u16* wstage = (u16*)carve((size_t)cH * cV * 2);      // K-packed bf16 weights
  u16* wqb = wstage;
  u16* wkb = wqb + (size_t)cH * cQD;
  u16* wvb = wkb + (size_t)cH * cKD;
  u16* wob = wvb + (size_t)cH * cKD;
  u16* wex = wob + (size_t)cQD * cH;   // one expert matrix at a time

  auto cvt = [&](const float* src, u16* dst, int Kd, int N) {
    const size_t n = (size_t)Kd * N / 4;
    cvt_pack_bf16<<<dim3((unsigned)((n + 255) / 256)), 256, 0, stream>>>(src, dst, Kd, N);
  };

  const dim3 blk256(256), blk128(128);
  const dim3 gQ(cQD / 128, cT / 64);
  const dim3 gKV(cKD / 128, cT / 64);
  const dim3 gH(cH / 128, cT / 64);
  const dim3 gI(cI / 128, cT / 64);
  const dim3 gV(cV / 128, cT / 64);

  embed_kernel<<<cT, blk256, 0, stream>>>(ids, embed, h);

  for (int l = 0; l < cL; l++) {
    cvt(Wq + (size_t)l * cH * cQD, wqb, cH, cQD);
    cvt(Wk + (size_t)l * cH * cKD, wkb, cH, cKD);
    cvt(Wv + (size_t)l * cH * cKD, wvb, cH, cKD);
    cvt(Wo + (size_t)l * cQD * cH, wob, cQD, cH);

    rmsnorm_kernel<<<cT, blk256, 0, stream>>>(h, ln1 + (size_t)l * cH, xb);
    gemm_bf16_lds<<<gQ, blk256, 0, stream>>>(xb, cH, wqb, cQD, nullptr, qb, cQD,
                                             cH, 4, nullptr, 0, nullptr, 0, 0);
    gemm_bf16_lds<<<gKV, blk256, 0, stream>>>(xb, cH, wkb, cKD, nullptr, kb, cKD,
                                              cH, 4, nullptr, 0, nullptr, 0, 0);
    gemm_bf16_lds<<<gKV, blk256, 0, stream>>>(xb, cH, wvb, cKD, nullptr, vtb, cKD,
                                              cH, 5, nullptr, 0, nullptr, 0, cT);
    attn_fused<<<cB * cNH * (cS / 64), blk128, 0, stream>>>(qb, kb, vtb, ob);
    gemm_bf16_lds<<<gH, blk256, 0, stream>>>(ob, cQD, wob, cH, h, nullptr, cH,
                                             cQD, 1, nullptr, 0, nullptr, 0, 0);

    // ---- MoE ----
    rmsnorm_kernel<<<cT, blk256, 0, stream>>>(h, ln2 + (size_t)l * cH, xb);
    gating_kernel<<<cT / 4, blk128, 0, stream>>>(xb, gate + (size_t)l * cH * cE, comb);
    for (int e = 0; e < cE; e++) {
      const size_t woff = ((size_t)l * cE + e) * (size_t)cH * cI;
      const size_t doff = ((size_t)l * cE + e) * (size_t)cI * cH;
      cvt(Wg + woff, wex, cH, cI);
      gemm_bf16_lds<<<gI, blk256, 0, stream>>>(xb, cH, wex, cI, gbuf, nullptr, cI,
                                               cH, 0, nullptr, 0, nullptr, 0, 0);
      cvt(Wu + woff, wex, cH, cI);
      gemm_bf16_lds<<<gI, blk256, 0, stream>>>(xb, cH, wex, cI, nullptr, gub, cI,
                                               cH, 2, gbuf, cI, nullptr, 0, 0);
      cvt(Wd + doff, wex, cI, cH);
      gemm_bf16_lds<<<gH, blk256, 0, stream>>>(gub, cI, wex, cH, h, nullptr, cH,
                                               cI, 3, nullptr, 0, comb + e, cE, 0);
    }
  }

  // ---- final norm + lm_head ----
  rmsnorm_kernel<<<cT, blk256, 0, stream>>>(h, norm_w, xb);
  cvt(lmh, wstage, cH, cV);
  gemm_bf16_lds<<<gV, blk256, 0, stream>>>(xb, cH, wstage, cV, out, nullptr, cV,
                                           cH, 0, nullptr, 0, nullptr, 0, 0);
}
